// HeteroGIN_16037407883348
// MI455X (gfx1250) — matrix operand order repaired and checked by reading, not compiled, not run on hardware
//
#include <hip/hip_runtime.h>
#include <hip/hip_bf16.h>

typedef __attribute__((ext_vector_type(16))) _Float16     v16h;
typedef __attribute__((ext_vector_type(8)))  _Float16     v8h;
typedef __attribute__((ext_vector_type(8)))  float        v8f;
typedef __attribute__((ext_vector_type(8)))  unsigned int v8u;

#define HID  128
#define OUTC 64
#define LN_EPS 1e-5f

// ---------------------------------------------------------------------------
// Pack a row-major f32 weight W[K][Ncols] into the per-lane WMMA B-fragment
// layout for V_WMMA_F32_16X16X32_F16 (ISA 7.12.2):
//   fragment f = nt*(K/32)+kb ; lane 0-15 hold K=0..7,16..23 (pairs in VGPR i),
//   lane 16-31 hold K=8..15,24..31.  Flat uint index = (f*32 + lane)*8 + i.
// ---------------------------------------------------------------------------
__global__ void k_pack_w(const float* __restrict__ W, unsigned int* __restrict__ out,
                         int K, int Ncols) {
  int u = blockIdx.x * blockDim.x + threadIdx.x;
  int KB = K >> 5;
  int total = (Ncols >> 4) * KB * 32 * 8;
  if (u >= total) return;
  int i    = u & 7;
  int lane = (u >> 3) & 31;
  int f    = u >> 8;
  int kb = f % KB, nt = f / KB;
  int khalf = lane >> 4;
  int col = nt * 16 + (lane & 15);
  int k0 = kb * 32 + khalf * 8 + 2 * i + (i >= 4 ? 8 : 0);
  union { _Float16 h[2]; unsigned int u32; } cv;
  cv.h[0] = (_Float16)W[(size_t)k0 * Ncols + col];
  cv.h[1] = (_Float16)W[(size_t)(k0 + 1) * Ncols + col];
  out[u] = cv.u32;
}

// ---------------------------------------------------------------------------
// Input projection: h[N,128] = x[N,16] @ W_in[16,128] + b_in.  Tiny GEMM
// (0.4 GFLOP) — one block per node, x row staged in LDS.
// ---------------------------------------------------------------------------
__global__ void k_input_proj(const float* __restrict__ x, const float* __restrict__ W,
                             const float* __restrict__ b, float* __restrict__ h) {
  __shared__ float xs[16];
  int i = blockIdx.x;
  if (threadIdx.x < 16) xs[threadIdx.x] = x[(size_t)i * 16 + threadIdx.x];
  __syncthreads();
  int j = threadIdx.x;
  float acc = b[j];
#pragma unroll
  for (int k = 0; k < 16; ++k) acc = fmaf(xs[k], W[k * HID + j], acc);
  h[(size_t)i * HID + j] = acc;
}

__global__ void k_zero(float* __restrict__ p, long long n) {
  long long t = (long long)blockIdx.x * blockDim.x + threadIdx.x;
  if (t < n) p[t] = 0.f;
}

// ---------------------------------------------------------------------------
// Edge scatter: agg[dst] += h[src].  One wave per edge, each lane moves a
// float4 (coalesced 512B gather) and does 4 f32 atomics.  This is the
// bandwidth-bound stage of the whole net (~1.6 GB for machine edges).
// ---------------------------------------------------------------------------
__global__ void k_scatter(const float* __restrict__ h, const int* __restrict__ src,
                          const int* __restrict__ dst, float* __restrict__ agg,
                          long long E) {
  long long t = (long long)blockIdx.x * blockDim.x + threadIdx.x;
  long long e = t >> 5;
  int l = (int)(t & 31);
  if (e >= E) return;
  int s = src[e], d = dst[e];
  float4 v = ((const float4*)(h + (size_t)s * HID))[l];
  float* base = agg + (size_t)d * HID + 4 * l;
  atomicAdd(base + 0, v.x);
  atomicAdd(base + 1, v.y);
  atomicAdd(base + 2, v.z);
  atomicAdd(base + 3, v.w);
}

// ---------------------------------------------------------------------------
// One GIN relation: out (+)= relu((h+agg) @ W1 + b1) @ W2 + b2
// 256 threads = 8 waves; each wave owns a 16-node tile and computes the full
// 16x128 output via 8(ntile) x 4(kblock) WMMAs per layer.  A-frags built in
// the 16-bit A layout; B-frags are coalesced 32B/lane loads from the packed
// weights (L2-resident).  ReLU intermediate staged in LDS as f16.
// ---------------------------------------------------------------------------
__global__ __launch_bounds__(256) void k_gin_mlp(
    const float* __restrict__ h, const float* __restrict__ agg,
    const unsigned int* __restrict__ W1p, const float* __restrict__ b1,
    const unsigned int* __restrict__ W2p, const float* __restrict__ b2,
    float* __restrict__ out, int accum, int n) {
  __shared__ __align__(16) _Float16 t1[8][16 * HID];  // 32 KB / workgroup
  int wave = threadIdx.x >> 5, lane = threadIdx.x & 31;
  int khalf = lane >> 4, lrow = lane & 15;
  int tileBase = (blockIdx.x * 8 + wave) * 16;
  int arow = tileBase + lrow;
  bool avalid = arow < n;
  const v8u* B1 = (const v8u*)W1p;
  const v8u* B2 = (const v8u*)W2p;

  // ---- A fragments of z = h + agg (f32 -> f16), 16-bit A layout ----
  v16h af[4];
#pragma unroll
  for (int kb = 0; kb < 4; ++kb) {
    int c0 = kb * 32 + khalf * 8;
#pragma unroll
    for (int j = 0; j < 16; ++j) {
      int col = c0 + (j < 8 ? j : 8 + j);  // elems 0..7 -> c0.., 8..15 -> c0+16..
      float v = 0.f;
      if (avalid) {
        size_t idx = (size_t)arow * HID + col;
        v = h[idx] + agg[idx];
      }
      af[kb][j] = (_Float16)v;
    }
  }

  // ---- layer 1: t1 = relu(z @ W1 + b1), staged to LDS as f16 ----
#pragma unroll
  for (int nt = 0; nt < 8; ++nt) {
    float bb = b1[nt * 16 + lrow];
    v8f acc = {bb, bb, bb, bb, bb, bb, bb, bb};
#pragma unroll
    for (int kb = 0; kb < 4; ++kb) {
      v8u raw = B1[(nt * 4 + kb) * 32 + lane];
      v16h bf = __builtin_bit_cast(v16h, raw);
      acc = __builtin_amdgcn_wmma_f32_16x16x32_f16(false, af[kb], false, bf,
                                                   (short)0, acc, false, false);
    }
#pragma unroll
    for (int r = 0; r < 8; ++r) {
      float v = acc[r] > 0.f ? acc[r] : 0.f;
      t1[wave][(r + 8 * khalf) * HID + nt * 16 + lrow] = (_Float16)v;
    }
  }

  // ---- layer 2 A-frags from LDS (already f16) ----
  v16h af2[4];
#pragma unroll
  for (int kb = 0; kb < 4; ++kb) {
    int c0 = kb * 32 + khalf * 8;
    v8h lo = *(const v8h*)&t1[wave][lrow * HID + c0];
    v8h hi = *(const v8h*)&t1[wave][lrow * HID + c0 + 16];
#pragma unroll
    for (int j = 0; j < 8; ++j) { af2[kb][j] = lo[j]; af2[kb][8 + j] = hi[j]; }
  }

  // ---- layer 2: out (+)= t1 @ W2 + b2 ----
#pragma unroll
  for (int nt = 0; nt < 8; ++nt) {
    float bb = b2[nt * 16 + lrow];
    v8f acc = {bb, bb, bb, bb, bb, bb, bb, bb};
#pragma unroll
    for (int kb = 0; kb < 4; ++kb) {
      v8u raw = B2[(nt * 4 + kb) * 32 + lane];
      v16h bf = __builtin_bit_cast(v16h, raw);
      acc = __builtin_amdgcn_wmma_f32_16x16x32_f16(false, af2[kb], false, bf,
                                                   (short)0, acc, false, false);
    }
#pragma unroll
    for (int r = 0; r < 8; ++r) {
      int m = tileBase + r + 8 * khalf;  // C/D layout: VGPR r -> row r + 8*(lane>=16)
      if (m < n) {
        size_t idx = (size_t)m * HID + nt * 16 + lrow;
        float prev = accum ? out[idx] : 0.f;
        out[idx] = prev + acc[r];
      }
    }
  }
}

// ---------------------------------------------------------------------------
// Fused LayerNorm + exact GELU + output projection to [N,64].
// Each lane holds half a row (the A-frag halves); one shfl_xor(16) finishes
// the 128-wide row reduction under wave32, then 4x4 WMMAs vs packed W_out.
// ---------------------------------------------------------------------------
__global__ __launch_bounds__(256) void k_final(
    const float* __restrict__ h2, const float* __restrict__ gamma,
    const float* __restrict__ beta, const unsigned int* __restrict__ Wop,
    const float* __restrict__ bo, float* __restrict__ out, int n) {
  int wave = threadIdx.x >> 5, lane = threadIdx.x & 31;
  int khalf = lane >> 4, lrow = lane & 15;
  int tileBase = (blockIdx.x * 8 + wave) * 16;
  int arow = tileBase + lrow;
  bool avalid = arow < n;

  float vals[4][16];
  float s = 0.f, ss = 0.f;
#pragma unroll
  for (int kb = 0; kb < 4; ++kb) {
    int c0 = kb * 32 + khalf * 8;
#pragma unroll
    for (int j = 0; j < 16; ++j) {
      int col = c0 + (j < 8 ? j : 8 + j);
      float v = avalid ? h2[(size_t)arow * HID + col] : 0.f;
      vals[kb][j] = v;
      s += v;
      ss += v * v;
    }
  }
  s  += __shfl_xor(s, 16);   // lane pair {l, l^16} together hold the full row
  ss += __shfl_xor(ss, 16);
  float mu   = s * (1.f / 128.f);
  float var  = ss * (1.f / 128.f) - mu * mu;
  float rstd = rsqrtf(var + LN_EPS);

  v16h af[4];
#pragma unroll
  for (int kb = 0; kb < 4; ++kb) {
    int c0 = kb * 32 + khalf * 8;
#pragma unroll
    for (int j = 0; j < 16; ++j) {
      int col = c0 + (j < 8 ? j : 8 + j);
      float v = (vals[kb][j] - mu) * rstd * gamma[col] + beta[col];
      v = 0.5f * v * (1.f + erff(v * 0.70710678118654752f));  // exact GELU
      af[kb][j] = (_Float16)v;
    }
  }

  const v8u* B = (const v8u*)Wop;
#pragma unroll
  for (int nt = 0; nt < 4; ++nt) {
    float bb = bo[nt * 16 + lrow];
    v8f acc = {bb, bb, bb, bb, bb, bb, bb, bb};
#pragma unroll
    for (int kb = 0; kb < 4; ++kb) {
      v8u raw = B[(nt * 4 + kb) * 32 + lane];
      v16h bf = __builtin_bit_cast(v16h, raw);
      acc = __builtin_amdgcn_wmma_f32_16x16x32_f16(false, af[kb], false, bf,
                                                   (short)0, acc, false, false);
    }
#pragma unroll
    for (int r = 0; r < 8; ++r) {
      int m = tileBase + r + 8 * khalf;
      if (m < n) out[(size_t)m * OUTC + nt * 16 + lrow] = acc[r];
    }
  }
}

// ---------------------------------------------------------------------------
extern "C" void kernel_launch(void* const* d_in, const int* in_sizes, int n_in,
                              void* d_out, int out_size, void* d_ws, size_t ws_size,
                              hipStream_t stream) {
  const float* x    = (const float*)d_in[0];
  const int*   je   = (const int*)d_in[1];
  const int*   me   = (const int*)d_in[2];
  const float* Win  = (const float*)d_in[3];
  const float* bin  = (const float*)d_in[4];
  const float* Wj1  = (const float*)d_in[5];
  const float* bj1  = (const float*)d_in[6];
  const float* Wj2  = (const float*)d_in[7];
  const float* bj2  = (const float*)d_in[8];
  const float* Wm1  = (const float*)d_in[9];
  const float* bm1  = (const float*)d_in[10];
  const float* Wm2  = (const float*)d_in[11];
  const float* bm2  = (const float*)d_in[12];
  const float* gam  = (const float*)d_in[13];
  const float* bet  = (const float*)d_in[14];
  const float* Wout = (const float*)d_in[15];
  const float* bout = (const float*)d_in[16];

  int N = in_sizes[0] / 16;
  long long Ej = in_sizes[1] / 2;
  long long Em = in_sizes[2] / 2;

  // workspace layout (f32 node buffers + packed f16 weights)
  float* h   = (float*)d_ws;
  float* agg = h + (size_t)N * HID;
  float* h2  = agg + (size_t)N * HID;
  unsigned int* w1j = (unsigned int*)(h2 + (size_t)N * HID);
  unsigned int* w2j = w1j + 8192;   // 128*128/2 packed uints each
  unsigned int* w1m = w2j + 8192;
  unsigned int* w2m = w1m + 8192;
  unsigned int* wop = w2m + 8192;   // 128*64/2 = 4096 uints

  // pack weights into WMMA B-fragment layout (tiny, L2-resident afterwards)
  k_pack_w<<<(8192 + 255) / 256, 256, 0, stream>>>(Wj1, w1j, 128, 128);
  k_pack_w<<<(8192 + 255) / 256, 256, 0, stream>>>(Wj2, w2j, 128, 128);
  k_pack_w<<<(8192 + 255) / 256, 256, 0, stream>>>(Wm1, w1m, 128, 128);
  k_pack_w<<<(8192 + 255) / 256, 256, 0, stream>>>(Wm2, w2m, 128, 128);
  k_pack_w<<<(4096 + 255) / 256, 256, 0, stream>>>(Wout, wop, 128, 64);

  // h = x @ W_in + b_in
  k_input_proj<<<N, 128, 0, stream>>>(x, Win, bin, h);

  long long nz = (long long)N * HID;
  int zb = (int)((nz + 255) / 256);
  int gemmBlocks = (N + 127) / 128;

  // relation: job (writes h2)
  k_zero<<<zb, 256, 0, stream>>>(agg, nz);
  k_scatter<<<(int)((Ej * 32 + 255) / 256), 256, 0, stream>>>(h, je, je + Ej, agg, Ej);
  k_gin_mlp<<<gemmBlocks, 256, 0, stream>>>(h, agg, w1j, bj1, w2j, bj2, h2, 0, N);

  // relation: machine (accumulates into h2)
  k_zero<<<zb, 256, 0, stream>>>(agg, nz);
  k_scatter<<<(int)((Em * 32 + 255) / 256), 256, 0, stream>>>(h, me, me + Em, agg, Em);
  k_gin_mlp<<<gemmBlocks, 256, 0, stream>>>(h, agg, w1m, bm1, w2m, bm2, h2, 1, N);

  // LayerNorm + GELU + output projection
  k_final<<<gemmBlocks, 256, 0, stream>>>(h2, gam, bet, wop, bout, (float*)d_out, N);
}
